// ConvBlock_44083544326214
// MI455X (gfx1250) — compile-verified, matrix operand order checked by reading, not run on hardware
//
#include <hip/hip_runtime.h>
#include <math.h>

// ---------------- problem constants (match reference) ----------------
#define N_NODES 50000
#define N_EDGES 400000
#define C_IN    16
#define C_OUT   32
#define KSPL    5              // kernel size per dim
#define W_ELEMS (125 * C_IN * C_OUT)   // 64000 floats = 256 KB, fits 320KB WGP LDS
#define BN_EPS  1e-5f

#define EDGE_BLOCKS 512
#define EDGE_THREADS 256

typedef __attribute__((ext_vector_type(16))) _Float16 v16h;
typedef __attribute__((ext_vector_type(8)))  float    v8f;
typedef __attribute__((ext_vector_type(4)))  int      v4i;

#define AS1 __attribute__((address_space(1)))
#define AS3 __attribute__((address_space(3)))

static __device__ __forceinline__ int imin(int a, int b) { return a < b ? a : b; }
static __device__ __forceinline__ int imax(int a, int b) { return a > b ? a : b; }

// =====================================================================
// Kernel 1: per-edge spline messages, accumulated with global atomics.
// Whole weight tensor (256KB) staged into LDS once per workgroup via the
// CDNA5 async global->LDS path (fallback: plain copy). lane = c_out so
// LDS reads are 32 consecutive dwords -> bank-conflict free.
// =====================================================================
__global__ void spline_edge_kernel(const float* __restrict__ x,
                                   const int*   __restrict__ ei,
                                   const float* __restrict__ ea,
                                   const float* __restrict__ w,
                                   float* __restrict__ agg,
                                   float* __restrict__ deg)
{
    extern __shared__ float ldsW[];
    const int tid = threadIdx.x;

    // ---- stage weights: 16000 float4 copies across 256 threads ----
    {
        const float4* src4 = (const float4*)w;
        float4* dst4 = (float4*)ldsW;
#if __has_builtin(__builtin_amdgcn_global_load_async_to_lds_b128)
        for (int i = tid; i < W_ELEMS / 4; i += EDGE_THREADS) {
            __builtin_amdgcn_global_load_async_to_lds_b128(
                (AS1 v4i*)(src4 + i), (AS3 v4i*)(dst4 + i), 0, 0);
        }
#if __has_builtin(__builtin_amdgcn_s_wait_asynccnt)
        __builtin_amdgcn_s_wait_asynccnt(0);
#else
        asm volatile("s_wait_asynccnt 0" ::: "memory");
#endif
#else
        for (int i = tid; i < W_ELEMS / 4; i += EDGE_THREADS) dst4[i] = src4[i];
#endif
    }
    __syncthreads();

    const int lane = tid & 31;      // lane == c_out
    const int wave = tid >> 5;      // 8 waves per block

    const int per = (N_EDGES + gridDim.x - 1) / gridDim.x;
    const int e0 = blockIdx.x * per;
    const int e1 = (e0 + per < N_EDGES) ? (e0 + per) : N_EDGES;

    for (int e = e0 + wave; e < e1; e += (EDGE_THREADS / 32)) {
        const int srcN = ei[e];
        const int dstN = ei[N_EDGES + e];

        // degree-1 open B-spline basis
        const float v0 = ea[3 * e + 0] * (float)(KSPL - 1);
        const float v1 = ea[3 * e + 1] * (float)(KSPL - 1);
        const float v2 = ea[3 * e + 2] * (float)(KSPL - 1);
        const int   l0 = (int)floorf(v0), l1 = (int)floorf(v1), l2 = (int)floorf(v2);
        const float f0 = v0 - floorf(v0), f1 = v1 - floorf(v1), f2 = v2 - floorf(v2);

        float xsv[C_IN];
#pragma unroll
        for (int c = 0; c < C_IN; ++c) xsv[c] = x[srcN * C_IN + c];  // wave-uniform -> broadcast

        float acc = 0.0f;
#pragma unroll
        for (int s = 0; s < 8; ++s) {
            const int b0 = s & 1, b1 = (s >> 1) & 1, b2 = (s >> 2) & 1;
            const int i0 = imin(imax(l0 + b0, 0), KSPL - 1);
            const int i1 = imin(imax(l1 + b1, 0), KSPL - 1);
            const int i2 = imin(imax(l2 + b2, 0), KSPL - 1);
            const float ws = (b0 ? f0 : 1.0f - f0) *
                             (b1 ? f1 : 1.0f - f1) *
                             (b2 ? f2 : 1.0f - f2);
            const int wi = i0 + KSPL * i1 + KSPL * KSPL * i2;
            const float* wp = ldsW + wi * (C_IN * C_OUT) + lane;
#pragma unroll
            for (int c = 0; c < C_IN; ++c)
                acc = fmaf(ws * xsv[c], wp[c * C_OUT], acc);
        }
        atomicAdd(&agg[dstN * C_OUT + lane], acc);
        if (lane == 0) atomicAdd(&deg[dstN], 1.0f);
    }
}

// =====================================================================
// Kernel 2: h = agg/deg + x@root + bias, ELU, + BN partial statistics.
// x@root done with v_wmma_f32_16x16x32_f16 (K padded 16->32 with zeros).
// One 16-node tile per wave; two WMMAs for the 32 output channels.
// A layout (16-bit, 16x32): lane%16 = M; lane/16 picks K-half:
//   elem j<8  -> K = j + 8*half ;  elem j>=8 -> K = 16 + (j-8) + 8*half
// D layout (f32 16x16): lane%16 = N; elem r -> M = r + 8*half.
// =====================================================================
__global__ void node_root_elu_kernel(const float* __restrict__ x,
                                     const float* __restrict__ root,
                                     const float* __restrict__ bias,
                                     const float* __restrict__ agg,
                                     const float* __restrict__ deg,
                                     float* __restrict__ h_out,
                                     float* __restrict__ chan_sums /* [0..31]=sum, [32..63]=sumsq */)
{
    __shared__ float bred[64];
    const int tid = threadIdx.x;
    if (tid < 64) bred[tid] = 0.0f;
    __syncthreads();

    const int lane = tid & 31;
    const int wave = tid >> 5;
    const int half = lane >> 4;
    const int li   = lane & 15;

    const int tiles = N_NODES / 16;  // 3125, exact
    const int tile = blockIdx.x * (blockDim.x >> 5) + wave;

    float psum0 = 0.f, psum1 = 0.f, psq0 = 0.f, psq1 = 0.f;

    if (tile < tiles) {
        const int nodeBase = tile * 16;
        v16h a, b0, b1;
#pragma unroll
        for (int j = 0; j < 16; ++j) {
            const int k = (j < 8 ? j : 16 + (j - 8)) + 8 * half;
            const bool live = (k < C_IN);
            a[j]  = live ? (_Float16)x[(nodeBase + li) * C_IN + k] : (_Float16)0.0f;
            b0[j] = live ? (_Float16)root[k * C_OUT + li]          : (_Float16)0.0f;
            b1[j] = live ? (_Float16)root[k * C_OUT + 16 + li]     : (_Float16)0.0f;
        }
        v8f c = {};
        v8f d0 = __builtin_amdgcn_wmma_f32_16x16x32_f16(false, a, false, b0, (short)0, c, false, false);
        v8f d1 = __builtin_amdgcn_wmma_f32_16x16x32_f16(false, a, false, b1, (short)0, c, false, false);

#pragma unroll
        for (int r = 0; r < 8; ++r) {
            const int node = nodeBase + r + 8 * half;
            const float inv = 1.0f / fmaxf(deg[node], 1.0f);
            float h0 = agg[node * C_OUT + li]      * inv + d0[r] + bias[li];
            float h1 = agg[node * C_OUT + 16 + li] * inv + d1[r] + bias[16 + li];
            h0 = (h0 > 0.0f) ? h0 : expm1f(h0);    // ELU (alpha=1)
            h1 = (h1 > 0.0f) ? h1 : expm1f(h1);
            h_out[node * C_OUT + li]      = h0;
            h_out[node * C_OUT + 16 + li] = h1;
            psum0 += h0; psq0 += h0 * h0;
            psum1 += h1; psq1 += h1 * h1;
        }
    }

    // block-level reduce -> 64 global atomics per block
    atomicAdd(&bred[li],       psum0);
    atomicAdd(&bred[16 + li],  psum1);
    atomicAdd(&bred[32 + li],  psq0);
    atomicAdd(&bred[48 + li],  psq1);
    __syncthreads();
    if (tid < 64) atomicAdd(&chan_sums[tid], bred[tid]);
}

// =====================================================================
// Kernel 3: BatchNorm apply (training-mode batch stats, biased variance),
// in-place over d_out.
// =====================================================================
__global__ void bn_apply_kernel(float* __restrict__ out,
                                const float* __restrict__ chan_sums,
                                const float* __restrict__ gamma,
                                const float* __restrict__ beta)
{
    const int i = blockIdx.x * blockDim.x + threadIdx.x;
    if (i >= N_NODES * C_OUT) return;
    const int c = i & (C_OUT - 1);
    const float invN = 1.0f / (float)N_NODES;
    const float mean = chan_sums[c] * invN;
    const float var  = chan_sums[32 + c] * invN - mean * mean;
    out[i] = (out[i] - mean) * rsqrtf(var + BN_EPS) * gamma[c] + beta[c];
}

// =====================================================================
extern "C" void kernel_launch(void* const* d_in, const int* in_sizes, int n_in,
                              void* d_out, int out_size, void* d_ws, size_t ws_size,
                              hipStream_t stream) {
    const float* x     = (const float*)d_in[0];
    const int*   ei    = (const int*)  d_in[1];
    const float* ea    = (const float*)d_in[2];
    const float* w     = (const float*)d_in[3];
    const float* root  = (const float*)d_in[4];
    const float* bias  = (const float*)d_in[5];
    const float* gamma = (const float*)d_in[6];
    const float* beta  = (const float*)d_in[7];
    float* out = (float*)d_out;

    // scratch layout: agg[N*32] | deg[N] | chan_sums[64]
    float* agg  = (float*)d_ws;
    float* deg  = agg + (size_t)N_NODES * C_OUT;
    float* sums = deg + N_NODES;
    const size_t zbytes = ((size_t)N_NODES * C_OUT + N_NODES + 64) * sizeof(float);
    (void)hipMemsetAsync(d_ws, 0, zbytes, stream);   // capture-safe, per-call determinism

    hipLaunchKernelGGL(spline_edge_kernel, dim3(EDGE_BLOCKS), dim3(EDGE_THREADS),
                       W_ELEMS * sizeof(float), stream, x, ei, ea, w, agg, deg);

    const int tiles = N_NODES / 16;                 // 3125
    const int nblocks = (tiles + 3) / 4;            // 4 waves/block, 128 threads
    hipLaunchKernelGGL(node_root_elu_kernel, dim3(nblocks), dim3(128), 0, stream,
                       x, root, bias, agg, deg, out, sums);

    const int total = N_NODES * C_OUT;
    hipLaunchKernelGGL(bn_apply_kernel, dim3((total + 255) / 256), dim3(256), 0, stream,
                       out, sums, gamma, beta);
}
